// Actor_71030169141505
// MI455X (gfx1250) — compile-verified, matrix operand order checked by reading, not run on hardware
//
#include <hip/hip_runtime.h>
#include <hip/hip_bf16.h>

// ---------------------------------------------------------------------------
// Types for CDNA5 WMMA (wave32): V_WMMA_F32_16X16X32_BF16
// ---------------------------------------------------------------------------
typedef __attribute__((ext_vector_type(16))) __bf16 bf16x16;
typedef __attribute__((ext_vector_type(8)))  __bf16 bf16x8;
typedef __attribute__((ext_vector_type(8)))  float  f32x8;
typedef __attribute__((ext_vector_type(4)))  float  f32x4;

#define VOCAB 32000
#define EMB   256
#define HID   512
#define BATCH 512
#define SEQ   256
#define VGRP  25      // V groups: 25 * 1280 = 32000
#define VG_COLS 1280  // cols per group (10 chunks of 128)

__device__ __forceinline__ f32x8 wmma_bf16(bf16x16 a, bf16x16 b, f32x8 c) {
  return __builtin_amdgcn_wmma_f32_16x16x32_bf16(
      /*neg_a=*/false, a, /*neg_b=*/false, b,
      /*c_mod=*/(short)0, c, /*reuse_a=*/false, /*reuse_b=*/false);
}

__device__ __forceinline__ bf16x16 cat16(bf16x8 lo, bf16x8 hi) {
  bf16x16 r;
#pragma unroll
  for (int i = 0; i < 8; ++i) { r[i] = lo[i]; r[i + 8] = hi[i]; }
  return r;
}

// Load a 16x32 bf16 A/B fragment run for this lane from a row-major matrix.
// Lane l&15 selects the row (A: M; B: N via w row), l>>4 selects the k-half.
// Per ISA layout: elements 0..7 at k = kbase + half*8, 8..15 at +16.
__device__ __forceinline__ bf16x16 load_frag_bf16(const __bf16* base, int ld,
                                                  int row, int kbase, int hf) {
  const __bf16* p = base + (size_t)row * ld + kbase + hf * 8;
  return cat16(*(const bf16x8*)p, *(const bf16x8*)(p + 16));
}

__device__ __forceinline__ bf16x8 cvt8_f32(const float* p) {
  f32x4 a = *(const f32x4*)p;
  f32x4 b = *(const f32x4*)(p + 4);
  bf16x8 r;
#pragma unroll
  for (int i = 0; i < 4; ++i) { r[i] = (__bf16)a[i]; r[i + 4] = (__bf16)b[i]; }
  return r;
}

// ---------------------------------------------------------------------------
// Hash RNG (deterministic; JAX bit-exactness is not reproducible anyway)
// ---------------------------------------------------------------------------
__device__ __forceinline__ unsigned mix32(unsigned x) {
  x ^= x >> 16; x *= 0x7feb352dU;
  x ^= x >> 15; x *= 0x846ca68bU;
  x ^= x >> 16; return x;
}
__device__ __forceinline__ float u01(unsigned a, unsigned b, unsigned c) {
  unsigned h = mix32(a * 0x8da6b343U ^ (b * 0xd8163841U + 0x165667b1U) ^ (c * 0xcb1ab31fU));
  return ((float)(h >> 8) + 0.5f) * (1.0f / 16777216.0f);
}
__device__ __forceinline__ float sigmoidf(float x) { return 1.0f / (1.0f + __expf(-x)); }

// ---------------------------------------------------------------------------
// One-shot: convert weights fp32 -> bf16 into workspace
// ---------------------------------------------------------------------------
__global__ void cvt_weights(const float* __restrict__ wih, const float* __restrict__ whh,
                            const float* __restrict__ wout,
                            __bf16* __restrict__ wih_b, __bf16* __restrict__ whh_b,
                            __bf16* __restrict__ wout_b) {
  size_t i = (size_t)blockIdx.x * blockDim.x + threadIdx.x;
  size_t stride = (size_t)gridDim.x * blockDim.x;
  for (size_t k = i; k < (size_t)3 * HID * EMB; k += stride) wih_b[k] = (__bf16)wih[k];
  for (size_t k = i; k < (size_t)3 * HID * HID; k += stride) whh_b[k] = (__bf16)whh[k];
  for (size_t k = i; k < (size_t)VOCAB * HID;  k += stride) wout_b[k] = (__bf16)wout[k];
}

// One-shot: h0 = 0, idx = 0 (start token) -> x0 = embedding[0]
__global__ void init_state(float* __restrict__ h32, __bf16* __restrict__ hbf,
                           int* __restrict__ idx) {
  size_t i = (size_t)blockIdx.x * blockDim.x + threadIdx.x;
  size_t stride = (size_t)gridDim.x * blockDim.x;
  for (size_t k = i; k < (size_t)BATCH * HID; k += stride) {
    h32[k] = 0.0f; hbf[k] = (__bf16)0.0f;
  }
  for (size_t k = i; k < BATCH; k += stride) idx[k] = 0;
}

// ---------------------------------------------------------------------------
// GRU step: one 16(batch) x 16(hidden) output tile per WG; 3 waves = r,z,n.
// gi = emb[idx] @ w_ih^T (E=256), gh = h @ w_hh^T (H=512), bf16 WMMA.
// ---------------------------------------------------------------------------
__global__ __launch_bounds__(96) void gru_step_kernel(
    const float* __restrict__ emb, const int* __restrict__ idx,
    const __bf16* __restrict__ wih_b, const __bf16* __restrict__ whh_b,
    const float* __restrict__ bih, const float* __restrict__ bhh,
    const float* __restrict__ h32_src, const __bf16* __restrict__ hbf_src,
    float* __restrict__ h32_dst, __bf16* __restrict__ hbf_dst) {
  __shared__ float xch[4][16][16];  // 0=R, 1=Z, 2=I_n, 3=H_n

  const int lane = threadIdx.x & 31;
  const int gate = threadIdx.x >> 5;  // 0=r, 1=z, 2=n
  const int hf   = lane >> 4;
  const int nr   = lane & 15;
  const int b0 = blockIdx.x * 16;
  const int c0 = blockIdx.y * 16;
  const int wrow = gate * HID + c0 + nr;  // B-matrix row in w_ih / w_hh

  // ---- gi accumulation over E (x gathered from embedding, fp32 -> bf16) ----
  const int arow = idx[b0 + nr];
  f32x8 acc_i = {0.f, 0.f, 0.f, 0.f, 0.f, 0.f, 0.f, 0.f};
#pragma unroll
  for (int kc = 0; kc < EMB / 32; ++kc) {
    const float* pa = emb + (size_t)arow * EMB + kc * 32 + hf * 8;
    bf16x16 a = cat16(cvt8_f32(pa), cvt8_f32(pa + 16));
    bf16x16 b = load_frag_bf16(wih_b, EMB, wrow, kc * 32, hf);
    acc_i = wmma_bf16(a, b, acc_i);
  }
  // ---- gh accumulation over H (h already staged as bf16) ----
  f32x8 acc_h = {0.f, 0.f, 0.f, 0.f, 0.f, 0.f, 0.f, 0.f};
#pragma unroll
  for (int kc = 0; kc < HID / 32; ++kc) {
    bf16x16 a = load_frag_bf16(hbf_src, HID, b0 + nr, kc * 32, hf);
    bf16x16 b = load_frag_bf16(whh_b, HID, wrow, kc * 32, hf);
    acc_h = wmma_bf16(a, b, acc_h);
  }

  // ---- gate epilogue: D[j] holds row (j + 8*hf), col nr ----
  const float bi = bih[gate * HID + c0 + nr];
  const float bh = bhh[gate * HID + c0 + nr];
#pragma unroll
  for (int j = 0; j < 8; ++j) {
    const int m = j + 8 * hf;
    const float vi = acc_i[j] + bi;
    const float vh = acc_h[j] + bh;
    if (gate == 0)      xch[0][m][nr] = sigmoidf(vi + vh);
    else if (gate == 1) xch[1][m][nr] = sigmoidf(vi + vh);
    else              { xch[2][m][nr] = vi; xch[3][m][nr] = vh; }
  }
  __syncthreads();

  for (int e = threadIdx.x; e < 256; e += 96) {
    const int m = e >> 4, n = e & 15;
    const float r = xch[0][m][n], z = xch[1][m][n];
    const float nn = tanhf(xch[2][m][n] + r * xch[3][m][n]);
    const size_t o = (size_t)(b0 + m) * HID + (c0 + n);
    const float hp = h32_src[o];
    const float hnew = (1.0f - z) * nn + z * hp;
    h32_dst[o] = hnew;
    hbf_dst[o] = (__bf16)hnew;
  }
}

// ---------------------------------------------------------------------------
// Logits + streaming log-softmax + Gumbel argmax partials.
// Grid (B/16, VGRP); 8 waves/WG, each wave owns a 16-col lane of the chunk.
// Never materializes logits: per-lane running (max, sumexp, best logit+g,
// best g) over the 1280 columns of this V group.
// ---------------------------------------------------------------------------
__global__ __launch_bounds__(256, 1) void logits_sample_partial(
    const __bf16* __restrict__ hbf, const __bf16* __restrict__ wout_b,
    const float* __restrict__ bout, float* __restrict__ part, int t) {
  const int lane = threadIdx.x & 31;
  const int wave = threadIdx.x >> 5;  // 0..7
  const int hf = lane >> 4;
  const int nr = lane & 15;
  const int b0 = blockIdx.x * 16;
  const int vg = blockIdx.y;
  const int vbase = vg * VG_COLS;

  // Preload all 16 A-fragments of the h tile (reused across 1280 columns).
  bf16x16 afrag[16];
#pragma unroll
  for (int kc = 0; kc < 16; ++kc)
    afrag[kc] = load_frag_bf16(hbf, HID, b0 + nr, kc * 32, hf);

  float rmax[8], rsum[8], bv[8], bvl[8], bg[8], bgl[8];
  int bvi[8], bgi[8];
#pragma unroll
  for (int j = 0; j < 8; ++j) {
    rmax[j] = -1e30f; rsum[j] = 0.0f;
    bv[j] = -1e30f; bvl[j] = 0.0f; bvi[j] = 0;
    bg[j] = -1e30f; bgl[j] = 0.0f; bgi[j] = 0;
  }

  for (int ch = 0; ch < VG_COLS / 128; ++ch) {
    const int v0 = vbase + ch * 128 + wave * 16;  // this wave's 16-col tile
    f32x8 acc = {0.f, 0.f, 0.f, 0.f, 0.f, 0.f, 0.f, 0.f};
#pragma unroll
    for (int kc = 0; kc < 16; ++kc) {
      bf16x16 b = load_frag_bf16(wout_b, HID, v0 + nr, kc * 32, hf);
      acc = wmma_bf16(afrag[kc], b, acc);
    }
    const int v = v0 + nr;  // this lane's global column
    const float bias = bout[v];
#pragma unroll
    for (int j = 0; j < 8; ++j) {
      const int brow = b0 + j + 8 * hf;
      const float logit = acc[j] + bias;
      // streaming log-sum-exp
      const float nm = fmaxf(rmax[j], logit);
      rsum[j] = rsum[j] * __expf(rmax[j] - nm) + __expf(logit - nm);
      rmax[j] = nm;
      // gumbel noise per (t, b, v)
      const float u = u01(0x6A09E667u + (unsigned)t, (unsigned)brow, (unsigned)v);
      const float g = -__logf(-__logf(u));
      const float tv = logit + g;
      if (tv > bv[j]) { bv[j] = tv; bvl[j] = logit; bvi[j] = v; }
      if (g > bg[j])  { bg[j] = g;  bgl[j] = logit; bgi[j] = v; }
    }
  }

  // Butterfly reduce across the 16 lanes sharing each row (stays in half).
#pragma unroll
  for (int m = 1; m < 16; m <<= 1) {
#pragma unroll
    for (int j = 0; j < 8; ++j) {
      const float om = __shfl_xor(rmax[j], m, 32);
      const float os = __shfl_xor(rsum[j], m, 32);
      const float M = fmaxf(rmax[j], om);
      rsum[j] = rsum[j] * __expf(rmax[j] - M) + os * __expf(om - M);
      rmax[j] = M;
      const float obv = __shfl_xor(bv[j], m, 32);
      const float obvl = __shfl_xor(bvl[j], m, 32);
      const int obvi = __shfl_xor(bvi[j], m, 32);
      if (obv > bv[j]) { bv[j] = obv; bvl[j] = obvl; bvi[j] = obvi; }
      const float obg = __shfl_xor(bg[j], m, 32);
      const float obgl = __shfl_xor(bgl[j], m, 32);
      const int obgi = __shfl_xor(bgi[j], m, 32);
      if (obg > bg[j]) { bg[j] = obg; bgl[j] = obgl; bgi[j] = obgi; }
    }
  }

  if (nr == 0) {  // lanes 0 and 16 hold rows 0..7 / 8..15
#pragma unroll
    for (int j = 0; j < 8; ++j) {
      const int brow = b0 + j + 8 * hf;
      float* p = part + (((size_t)brow * VGRP + vg) * 8 + wave) * 8;
      p[0] = rmax[j]; p[1] = rsum[j];
      p[2] = bv[j];   p[3] = bvl[j]; p[4] = __int_as_float(bvi[j]);
      p[5] = bg[j];   p[6] = bgl[j]; p[7] = __int_as_float(bgi[j]);
    }
  }
}

// ---------------------------------------------------------------------------
// Finalize: combine partials, epsilon-greedy pick, write samples/lp/idx.
// ---------------------------------------------------------------------------
__global__ void finalize_sample(const float* __restrict__ part,
                                float* __restrict__ out_samples,
                                float* __restrict__ out_lps,
                                int* __restrict__ idx, int t) {
  const int b = blockIdx.x * blockDim.x + threadIdx.x;
  if (b >= BATCH) return;
  float M = -1e30f, Ssum = 0.0f;
  float bv = -1e30f, bvl = 0.0f, bg = -1e30f, bgl = 0.0f;
  int bvi = 0, bgi = 0;
  for (int g = 0; g < VGRP; ++g) {
    for (int w = 0; w < 8; ++w) {
      const float* p = part + (((size_t)b * VGRP + g) * 8 + w) * 8;
      const float m2 = p[0], s2 = p[1];
      const float nm = fmaxf(M, m2);
      Ssum = Ssum * __expf(M - nm) + s2 * __expf(m2 - nm);
      M = nm;
      if (p[2] > bv) { bv = p[2]; bvl = p[3]; bvi = __float_as_int(p[4]); }
      if (p[5] > bg) { bg = p[5]; bgl = p[6]; bgi = __float_as_int(p[7]); }
    }
  }
  const float lse = M + __logf(Ssum);
  const float eps = 0.05f + 0.95f * __expf(-4.0f * (float)t / 10000.0f);
  const float ue = u01(0x9E3779B9u + (unsigned)t, (unsigned)b, 0xC0FFEEu);
  int sampled; float lp;
  if (eps >= ue) { sampled = bgi; lp = bgl - lse; }  // whole row masked: argmax(g)
  else           { sampled = bvi; lp = bvl - lse; }  // argmax(logit + g)
  idx[b] = sampled;
  out_samples[(size_t)b * SEQ + t] = (float)sampled;
  out_lps[(size_t)b * SEQ + t] = lp;
}

// ---------------------------------------------------------------------------
// Host: workspace layout + 256-step launch sequence (graph-capture safe)
// ---------------------------------------------------------------------------
extern "C" void kernel_launch(void* const* d_in, const int* in_sizes, int n_in,
                              void* d_out, int out_size, void* d_ws, size_t ws_size,
                              hipStream_t stream) {
  const float* emb  = (const float*)d_in[0];
  const float* wih  = (const float*)d_in[1];
  const float* whh  = (const float*)d_in[2];
  const float* bih  = (const float*)d_in[3];
  const float* bhh  = (const float*)d_in[4];
  const float* wout = (const float*)d_in[5];
  const float* bout = (const float*)d_in[6];

  char* ws = (char*)d_ws;
  size_t off = 0;
  auto take = [&](size_t bytes) -> void* {
    void* p = ws + off;
    off += (bytes + 255) & ~(size_t)255;
    return p;
  };
  __bf16* wih_b  = (__bf16*)take((size_t)3 * HID * EMB * 2);
  __bf16* whh_b  = (__bf16*)take((size_t)3 * HID * HID * 2);
  __bf16* wout_b = (__bf16*)take((size_t)VOCAB * HID * 2);
  float*  h32    = (float*) take((size_t)2 * BATCH * HID * 4);
  __bf16* hbf    = (__bf16*)take((size_t)2 * BATCH * HID * 2);
  int*    idxb   = (int*)   take((size_t)BATCH * 4);
  float*  part   = (float*) take((size_t)BATCH * VGRP * 8 * 8 * 4);
  if (off > ws_size) return;  // workspace too small: do nothing (deterministic)

  float* out_samples = (float*)d_out;
  float* out_lps     = (float*)d_out + (size_t)BATCH * SEQ;

  cvt_weights<<<2048, 256, 0, stream>>>(wih, whh, wout, wih_b, whh_b, wout_b);
  init_state<<<1024, 256, 0, stream>>>(h32, hbf, idxb);

  for (int t = 0; t < SEQ; ++t) {
    const int src = t & 1;
    const int dst = src ^ 1;
    gru_step_kernel<<<dim3(BATCH / 16, HID / 16), 96, 0, stream>>>(
        emb, idxb, wih_b, whh_b, bih, bhh,
        h32 + (size_t)src * BATCH * HID, hbf + (size_t)src * BATCH * HID,
        h32 + (size_t)dst * BATCH * HID, hbf + (size_t)dst * BATCH * HID);
    logits_sample_partial<<<dim3(BATCH / 16, VGRP), 256, 0, stream>>>(
        hbf + (size_t)dst * BATCH * HID, wout_b, bout, part, t);
    finalize_sample<<<2, 256, 0, stream>>>(part, out_samples, out_lps, idxb, t);
  }
}
